// HopfieldAttention_46351287058950
// MI455X (gfx1250) — matrix operand order repaired and checked by reading, not compiled, not run on hardware
//
#include <hip/hip_runtime.h>
#include <hip/hip_bf16.h>

// ---------------------------------------------------------------------------
// Hopfield attention for MI455X (gfx1250): bf16 WMMA, TDM + async-to-LDS.
// ---------------------------------------------------------------------------

typedef __attribute__((ext_vector_type(16))) __bf16 v16bf;
typedef __attribute__((ext_vector_type(8)))  __bf16 v8bf;
typedef __attribute__((ext_vector_type(8)))  float  v8f;

static constexpr int Bn = 8, Sn = 1024, Dn = 512, Hn = 8, DKn = 64;
static constexpr int Mn = Bn * Sn;                 // 8192 tokens
static constexpr size_t CH = (size_t)Mn * Dn * 2;  // bytes of one bf16 (8192x512) buffer
static constexpr size_t WB = (size_t)Dn * Dn * 2;  // bytes of one bf16 (512x512) weight

// ---- WMMA fragment helpers (contiguous k = 16*half + e for A and B) ------

// A (16x32 bf16): lane half = lane>>4, m = lane&15.
// elements e -> K = 16*half + e (one contiguous 32B)
__device__ __forceinline__ v16bf ldA(const __bf16* base, int ld, int row0, int k0, int lane) {
    int half = lane >> 4, m = lane & 15;
    const __bf16* p = base + (size_t)(row0 + m) * ld + k0 + 16 * half;
    return *(const v16bf*)(p);
}

// B (32x16 bf16) read from a TRANSPOSED row-major array T[n][k]:
// lane holds column n = lane&15, elements e -> K = 16*half + e (one contiguous 32B)
__device__ __forceinline__ v16bf ldB(const __bf16* T, int ld, int n0, int k0, int lane) {
    int half = lane >> 4, n = lane & 15;
    return *(const v16bf*)(T + (size_t)(n0 + n) * ld + k0 + 16 * half);
}

__device__ __forceinline__ v8f wmma_bf16(v16bf a, v16bf b, v8f c) {
    return __builtin_amdgcn_wmma_f32_16x16x32_bf16(false, a, false, b, (short)0, c, false, false);
}

// ---- CDNA5 async copy: global -> LDS, 16 bytes per lane --------------------
__device__ __forceinline__ void async_b128(void* lds_dst, const void* gsrc) {
    unsigned ldsa = (unsigned)(unsigned long long)lds_dst;  // low 32 bits = LDS offset
    asm volatile("global_load_async_to_lds_b128 %0, %1, off"
                 :: "v"(ldsa), "v"((unsigned long long)gsrc) : "memory");
}
__device__ __forceinline__ void wait_async0() {
#if __has_builtin(__builtin_amdgcn_s_wait_asynccnt)
    __builtin_amdgcn_s_wait_asynccnt(0);
#else
    asm volatile("s_wait_asynccnt 0" ::: "memory");
#endif
}
__device__ __forceinline__ void wait_async4() {
#if __has_builtin(__builtin_amdgcn_s_wait_asynccnt)
    __builtin_amdgcn_s_wait_asynccnt(4);
#else
    asm volatile("s_wait_asynccnt 4" ::: "memory");
#endif
}

// ---- CDNA5 Tensor Data Mover: one 2D bf16 tile (64 x 64) -> LDS ------------
#if defined(__gfx1250__) && __has_builtin(__builtin_amdgcn_tensor_load_to_lds)
#define HOP_HAS_TDM 1
typedef __attribute__((ext_vector_type(4))) unsigned tdm_u4;
typedef __attribute__((ext_vector_type(8))) int      tdm_i8;
typedef __attribute__((ext_vector_type(4))) int      tdm_i4;

// D# per CDNA5 ISA: count=1, type=2 ("image"),
// data_size=2B, tensor_dim0=tensor_dim1=64, tile 64x64, row stride = stride0.
// clang-23 toolchain: 6-arg builtin (g0 u32x4, g1 i32x8, i32x4, i32x4, i32x8, cpol).
__device__ __forceinline__ void tdm_load_2d(void* lds_dst, const void* gsrc, unsigned stride0) {
    unsigned long long ga = (unsigned long long)gsrc;
    tdm_u4 g0;
    g0[0] = 1u;                                               // count=1 (user D#)
    g0[1] = (unsigned)(unsigned long long)lds_dst;            // lds_addr
    g0[2] = (unsigned)(ga & 0xFFFFFFFFu);                     // global_addr[31:0]
    g0[3] = (unsigned)((ga >> 32) & 0x1FFFFFFu) | (2u << 30); // global_addr[56:32], type=2
    tdm_i8 g1;
    g1[0] = 1 << 16;        // workgroup_mask=0 (not in cluster); data_size=1 (2 bytes)
    g1[1] = 64 << 16;       // tensor_dim0[15:0] = 64
    g1[2] = 64 << 16;       // tensor_dim0[31:16]=0 | tensor_dim1[15:0] = 64
    g1[3] = 64 << 16;       // tensor_dim1[31:16]=0 | tile_dim0 = 64
    g1[4] = 64;             // tile_dim1 = 64 | tile_dim2 = 0
    g1[5] = (int)stride0;   // tensor_dim0_stride[31:0] (elements)
    g1[6] = 0;              // stride0[47:32]=0 | stride1[15:0]=0
    g1[7] = 0;              // stride1[47:16]=0
    tdm_i4 gz4 = {0, 0, 0, 0};                 // groups 2/3 unused for 2D tiles
    tdm_i8 gz8 = {0, 0, 0, 0, 0, 0, 0, 0};
    __builtin_amdgcn_tensor_load_to_lds(g0, g1, gz4, gz4, gz8, 0);
}
#else
#define HOP_HAS_TDM 0
#endif

// fallback per-lane async staging of one K/V^T chunk pair
__device__ __forceinline__ void stage_async_chunk(__bf16* kd, __bf16* vd,
        const __bf16* kbase, const __bf16* vTbase, int ch, int t) {
    int rr = t >> 2, c0 = (t & 3) * 16;
    const __bf16* gk = kbase + (size_t)(ch * 64 + rr) * 64 + c0;
    async_b128(kd + rr * 64 + c0,     gk);
    async_b128(kd + rr * 64 + c0 + 8, gk + 8);
    const __bf16* gv = vTbase + (size_t)rr * Sn + ch * 64 + c0;
    async_b128(vd + rr * 64 + c0,     gv);
    async_b128(vd + rr * 64 + c0 + 8, gv + 8);
}

// ---------------------------------------------------------------------------
__global__ void hop_init_kernel(float* diffs) {
    if (threadIdx.x < 4) diffs[threadIdx.x] = 0.0f;
}

__global__ void hop_cast_kernel(const float* __restrict__ in, __bf16* __restrict__ out, int n) {
    int i = blockIdx.x * blockDim.x + threadIdx.x;
    if (i < n) out[i] = (__bf16)in[i];
}

// cast + transpose a 512x512 f32 weight into bf16 Wt[n][k]
__global__ void hop_prep_w_kernel(const float* __restrict__ W, __bf16* __restrict__ Wt) {
    int i = blockIdx.x * blockDim.x + threadIdx.x;  // i = n*512 + k
    int n = i >> 9, k = i & 511;
    Wt[i] = (__bf16)W[(size_t)k * Dn + n];
}

// ---------------------------------------------------------------------------
// GEMM: C[64x64 tile] = A_bf16(8192x512) @ Wt_bf16(n,k)^T + bias
// LDS-free mainloop: WMMA fragments loaded directly from global (L2-resident).
// mode 0: LayerNorm over the 64-col head dim; mode 1: plain; mode 2: f32 out.
__global__ __launch_bounds__(256) void hop_gemm_kernel(
        const __bf16* __restrict__ A, const __bf16* __restrict__ Wt,
        const float* __restrict__ bias, const float* __restrict__ gamma,
        const float* __restrict__ beta, __bf16* __restrict__ out_bf,
        __bf16* __restrict__ out_t, float* __restrict__ out_f32, int mode) {
    __shared__ float lds_out[64][68];

    const int t = threadIdx.x, w = t >> 5, lane = t & 31;
    const int half = lane >> 4, lane15 = lane & 15;
    const int M0 = blockIdx.x * 64, N0 = blockIdx.y * 64;
    const int ni = w & 3;            // column subtile
    const int mrow0 = (w >> 2) * 32; // this wave covers rows [mrow0, mrow0+32)

    v8f acc0 = {}, acc1 = {};
    #pragma unroll 4
    for (int kt = 0; kt < 16; ++kt) {
        v16bf b  = ldB(Wt, Dn, N0 + 16 * ni, kt * 32, lane);
        v16bf a0 = ldA(A, Dn, M0 + mrow0, kt * 32, lane);
        v16bf a1 = ldA(A, Dn, M0 + mrow0 + 16, kt * 32, lane);
        acc0 = wmma_bf16(a0, b, acc0);
        acc1 = wmma_bf16(a1, b, acc1);
    }

    const int col = 16 * ni + lane15;
    const float bv = bias[N0 + col];
    #pragma unroll
    for (int r = 0; r < 8; ++r) {
        lds_out[mrow0 + r + 8 * half][col]      = acc0[r] + bv;
        lds_out[mrow0 + 16 + r + 8 * half][col] = acc1[r] + bv;
    }
    __syncthreads();

    if (mode == 0 && t < 64) {   // LayerNorm in place
        float* row = lds_out[t];
        float mu = 0.f;
        for (int d = 0; d < 64; ++d) mu += row[d];
        mu *= (1.0f / 64.0f);
        float var = 0.f;
        for (int d = 0; d < 64; ++d) { float c = row[d] - mu; var += c * c; }
        var *= (1.0f / 64.0f);
        const float rstd = rsqrtf(var + 1e-5f);
        for (int d = 0; d < 64; ++d)
            row[d] = (row[d] - mu) * rstd * gamma[d] + beta[d];
    }
    __syncthreads();

    if (t < 64) {
        const int m = M0 + t;
        const float* row = lds_out[t];
        if (mode == 2) {
            float* o = out_f32 + (size_t)m * Dn + N0;
            for (int d = 0; d < 64; ++d) o[d] = row[d];
        } else if (out_bf) {
            const int b = m >> 10, s = m & 1023, h = N0 >> 6;
            __bf16* o = out_bf + ((size_t)(b * Hn + h) * Sn + s) * DKn;
            for (int d = 0; d < 64; ++d) o[d] = (__bf16)row[d];
        }
        if (out_t) {  // transposed (B,H,64,S): thread owns dim d = t
            const int b = M0 >> 10, h = N0 >> 6, smb = M0 & 1023;
            __bf16* o = out_t + ((size_t)((b * Hn + h) * DKn + t)) * Sn + smb;
            for (int s = 0; s < 64; ++s) o[s] = (__bf16)lds_out[s][t];
        }
    }
}

// ---------------------------------------------------------------------------
// Flash-style Hopfield attention step: out = softmax(q k^T / 8) @ val
// q/k: bf16 (B,H,S,64); valT: bf16 (B,H,64,S).
// K / V^T chunks double-buffered via TDM (fallback: per-lane async-to-LDS);
// Q tile via global_load_async_to_lds_b128.
__global__ __launch_bounds__(256) void hop_attn_kernel(
        const __bf16* __restrict__ q, const __bf16* __restrict__ k,
        const __bf16* __restrict__ valT, __bf16* __restrict__ out,
        float* __restrict__ diffsq, int final_mode) {
    __shared__ __align__(32) __bf16 q_lds[128 * 64];      // [q][d]        16 KB
    __shared__ __align__(32) __bf16 k_lds[2][64 * 64];    // [key][d]    2x 8 KB
    __shared__ __align__(32) __bf16 vT_lds[2][64 * 64];   // [d][key]    2x 8 KB
    __shared__ __align__(32) __bf16 p_lds[8][16 * 64];    // per-wave P    16 KB

    const int t = threadIdx.x, w = t >> 5, lane = t & 31;
    const int half = lane >> 4, lane15 = lane & 15;
    const int bh = blockIdx.y;
    const int q0 = blockIdx.x * 128;

    const __bf16* qbase  = q + ((size_t)bh * Sn + q0) * DKn;
    const __bf16* kbase  = k + (size_t)bh * Sn * DKn;
    const __bf16* vTbase = valT + (size_t)bh * DKn * Sn;

    // async-load Q tile (128x64 = 16KB): 64B per thread
    {
        int r = t >> 1, c0 = (t & 1) * 32;
        #pragma unroll
        for (int j = 0; j < 4; ++j)
            async_b128(q_lds + r * 64 + c0 + 8 * j, qbase + r * 64 + c0 + 8 * j);
    }
    // stage chunk 0
#if HOP_HAS_TDM
    if (t < 32) {
        tdm_load_2d(k_lds[0], kbase, 64);
        tdm_load_2d(vT_lds[0], vTbase, Sn);
    }
#else
    stage_async_chunk(k_lds[0], vT_lds[0], kbase, vTbase, 0, t);
#endif

    float mi[8], li[8];
    v8f o[4];
    #pragma unroll
    for (int r = 0; r < 8; ++r) { mi[r] = -3.0e38f; li[r] = 0.f; }
    #pragma unroll
    for (int n = 0; n < 4; ++n) o[n] = (v8f){};

    for (int ch = 0; ch < 16; ++ch) {
        const int cur = ch & 1, nxt = cur ^ 1;
        __syncthreads();  // all waves done reading buffer 'nxt' from chunk ch-1
#if HOP_HAS_TDM
        if (t < 32) {     // TDM issue is per-wave; wave 0 drives the data mover
            if (ch < 15) {
                tdm_load_2d(k_lds[nxt], kbase + (size_t)(ch + 1) * 64 * 64, 64);
                tdm_load_2d(vT_lds[nxt], vTbase + (ch + 1) * 64, Sn);
                __builtin_amdgcn_s_wait_tensorcnt(2);  // chunk ch retired (in-order)
            } else {
                __builtin_amdgcn_s_wait_tensorcnt(0);
            }
        }
        if (ch == 0) wait_async0();  // Q tile arrival
#else
        if (ch < 15) {
            stage_async_chunk(k_lds[nxt], vT_lds[nxt], kbase, vTbase, ch + 1, t);
            wait_async4();           // chunk ch (+Q on first iter) retired, ch+1 in flight
        } else {
            wait_async0();
        }
#endif
        __syncthreads();

        // scores: 16 q-rows x 64 keys, scaled by 1/sqrt(64)
        v8f sc[4];
        #pragma unroll
        for (int n = 0; n < 4; ++n) {
            v8f s = {};
            #pragma unroll
            for (int st = 0; st < 2; ++st) {
                v16bf a = ldA(q_lds, 64, 16 * w, 32 * st, lane);
                v16bf b = ldB(k_lds[cur], 64, 16 * n, 32 * st, lane);
                s = wmma_bf16(a, b, s);
            }
            #pragma unroll
            for (int r = 0; r < 8; ++r) s[r] *= 0.125f;
            sc[n] = s;
        }

        // online softmax per row (rows differ by (r, half); reduce over 16 lanes)
        #pragma unroll
        for (int r = 0; r < 8; ++r) {
            float mx = fmaxf(fmaxf(sc[0][r], sc[1][r]), fmaxf(sc[2][r], sc[3][r]));
            #pragma unroll
            for (int msk = 1; msk < 16; msk <<= 1) mx = fmaxf(mx, __shfl_xor(mx, msk, 32));
            const float mnew = fmaxf(mi[r], mx);
            const float f = __expf(mi[r] - mnew);
            float rs = 0.f;
            #pragma unroll
            for (int n = 0; n < 4; ++n) {
                const float p = __expf(sc[n][r] - mnew);
                rs += p;
                p_lds[w][(r + 8 * half) * 64 + 16 * n + lane15] = (__bf16)p;
            }
            #pragma unroll
            for (int msk = 1; msk < 16; msk <<= 1) rs += __shfl_xor(rs, msk, 32);
            li[r] = li[r] * f + rs;
            mi[r] = mnew;
            #pragma unroll
            for (int n = 0; n < 4; ++n) o[n][r] *= f;
        }

        // o += P @ V_chunk   (P: 16x64 over keys; V^T staged as [d][key])
        #pragma unroll
        for (int nd = 0; nd < 4; ++nd) {
            #pragma unroll
            for (int st = 0; st < 2; ++st) {
                v16bf a = ldA(&p_lds[w][0], 64, 0, 32 * st, lane);
                v16bf b = ldB(vT_lds[cur], 64, 16 * nd, 32 * st, lane);
                o[nd] = wmma_bf16(a, b, o[nd]);
            }
        }
    }

    // epilogue: normalize, write, accumulate diff^2 vs old q
    float dsum = 0.f;
    #pragma unroll
    for (int r = 0; r < 8; ++r) {
        const float inv = 1.0f / li[r];
        const int qrow = 16 * w + r + 8 * half;  // within 128-query block
        #pragma unroll
        for (int nd = 0; nd < 4; ++nd) {
            const float v = o[nd][r] * inv;
            const int dcol = 16 * nd + lane15;
            if (final_mode == 0) {
                const float qo = (float)q_lds[qrow * 64 + dcol];
                const float dd = v - qo;
                dsum += dd * dd;
                out[((size_t)bh * Sn + q0 + qrow) * DKn + dcol] = (__bf16)v;
            } else {
                const int b = bh >> 3, h = bh & 7;
                out[((size_t)b * Sn + q0 + qrow) * Dn + h * DKn + dcol] = (__bf16)v;
            }
        }
    }
    if (final_mode == 0) {
        #pragma unroll
        for (int msk = 1; msk < 32; msk <<= 1) dsum += __shfl_xor(dsum, msk, 32);
        if (lane == 0) atomicAdd(diffsq, dsum);
    }
}

// ---------------------------------------------------------------------------
// Eps-gated commit: q <- new_q iff ||new_q - q|| > 1e-4  (matches reference)
__global__ void hop_commit_kernel(uint4* __restrict__ dst, const uint4* __restrict__ src,
                                  const float* __restrict__ diffsq, int n16) {
    if (sqrtf(*diffsq) <= 1e-4f) return;
    int i = blockIdx.x * blockDim.x + threadIdx.x;
    if (i < n16) dst[i] = src[i];
}

// ---------------------------------------------------------------------------
extern "C" void kernel_launch(void* const* d_in, const int* in_sizes, int n_in,
                              void* d_out, int out_size, void* d_ws, size_t ws_size,
                              hipStream_t stream) {
    (void)in_sizes; (void)n_in; (void)out_size; (void)ws_size;
    const float* x     = (const float*)d_in[0];
    const float* Wq    = (const float*)d_in[1];
    const float* bq    = (const float*)d_in[2];
    const float* Wk    = (const float*)d_in[3];
    const float* bk    = (const float*)d_in[4];
    const float* Wv    = (const float*)d_in[5];
    const float* bv    = (const float*)d_in[6];
    const float* Wo    = (const float*)d_in[7];
    const float* bo    = (const float*)d_in[8];
    const float* gamma = (const float*)d_in[9];
    const float* beta  = (const float*)d_in[10];
    float* out = (float*)d_out;

    char* ws = (char*)d_ws;
    __bf16* xbf = (__bf16*)(ws + 0 * CH);
    __bf16* qb  = (__bf16*)(ws + 1 * CH);
    __bf16* kb  = (__bf16*)(ws + 2 * CH);
    __bf16* kT  = (__bf16*)(ws + 3 * CH);
    __bf16* vT  = (__bf16*)(ws + 4 * CH);
    __bf16* qn  = (__bf16*)(ws + 5 * CH);
    __bf16* ao  = (__bf16*)(ws + 6 * CH);
    __bf16* wqT = (__bf16*)(ws + 7 * CH);
    __bf16* wkT = (__bf16*)(ws + 7 * CH + 1 * WB);
    __bf16* wvT = (__bf16*)(ws + 7 * CH + 2 * WB);
    __bf16* woT = (__bf16*)(ws + 7 * CH + 3 * WB);
    float* diffs = (float*)(ws + 7 * CH + 4 * WB);

    hop_init_kernel<<<1, 32, 0, stream>>>(diffs);
    hop_cast_kernel<<<(Mn * Dn) / 256, 256, 0, stream>>>(x, xbf, Mn * Dn);
    hop_prep_w_kernel<<<(Dn * Dn) / 256, 256, 0, stream>>>(Wq, wqT);
    hop_prep_w_kernel<<<(Dn * Dn) / 256, 256, 0, stream>>>(Wk, wkT);
    hop_prep_w_kernel<<<(Dn * Dn) / 256, 256, 0, stream>>>(Wv, wvT);
    hop_prep_w_kernel<<<(Dn * Dn) / 256, 256, 0, stream>>>(Wo, woT);

    const dim3 gg(Mn / 64, Dn / 64);
    hop_gemm_kernel<<<gg, 256, 0, stream>>>(xbf, wqT, bq, gamma, beta, qb, nullptr, nullptr, 0);
    hop_gemm_kernel<<<gg, 256, 0, stream>>>(xbf, wkT, bk, gamma, beta, kb, kT, nullptr, 0);
    hop_gemm_kernel<<<gg, 256, 0, stream>>>(xbf, wvT, bv, gamma, beta, nullptr, vT, nullptr, 1);

    const dim3 ga(Sn / 128, Bn * Hn);
    const int n16 = (int)(CH / 16);
    for (int it = 0; it < 3; ++it) {
        hop_attn_kernel<<<ga, 256, 0, stream>>>(qb, kb, kT, qn, diffs + it, 0);
        hop_commit_kernel<<<(n16 + 255) / 256, 256, 0, stream>>>(
            (uint4*)qb, (const uint4*)qn, diffs + it, n16);
    }
    hop_attn_kernel<<<ga, 256, 0, stream>>>(qb, kb, vT, ao, diffs + 3, 1);

    hop_gemm_kernel<<<gg, 256, 0, stream>>>(ao, woT, bo, gamma, beta, nullptr, nullptr, out, 2);
}
